// UncertaintyMPNN_38551626449493
// MI455X (gfx1250) — matrix operand order repaired and checked by reading, not compiled, not run on hardware
//
#include <hip/hip_runtime.h>
#include <math.h>

// ---------------------------------------------------------------------------
// CDNA5 (gfx1250) WMMA helpers
// ---------------------------------------------------------------------------
typedef __attribute__((ext_vector_type(16))) __bf16 v16bf;
typedef __attribute__((ext_vector_type(8)))  float  v8f;

union AFrag { v16bf v; uint4 q[2]; unsigned short s[16]; };
union CFrag { v8f  v; float f[8]; };

__device__ __forceinline__ unsigned short f2bf(float f) {
  unsigned u = __float_as_uint(f);
  u += 0x7FFFu + ((u >> 16) & 1u);          // round-to-nearest-even
  return (unsigned short)(u >> 16);
}

__device__ __forceinline__ unsigned fenc(float f) {   // order-preserving f32->u32
  unsigned u = __float_as_uint(f);
  return (u & 0x80000000u) ? ~u : (u | 0x80000000u);
}
__device__ __forceinline__ float fdec(unsigned u) {
  unsigned v = (u & 0x80000000u) ? (u & 0x7fffffffu) : ~u;
  return __uint_as_float(v);
}
// host-side constant: fenc(-inf) = ~0xFF800000
#define FENC_NEG_INF 0x007FFFFFu

// ---------------------------------------------------------------------------
// CDNA5 async global->LDS copy (ASYNCcnt) -- CDNA5 ISA
// lds_off: LDS byte offset (low 32 bits of flat shared address)
// ---------------------------------------------------------------------------
__device__ __forceinline__ void async_ld32(unsigned lds_off, const unsigned short* g) {
  asm volatile("global_load_async_to_lds_b32 %0, %1, off"
               :: "v"(lds_off), "v"(g) : "memory");
}
__device__ __forceinline__ void wait_async0() {
  asm volatile("s_wait_asynccnt 0x0" ::: "memory");
}
__device__ __forceinline__ unsigned lds_off_of(const void* p) {
  return (unsigned)(size_t)p;   // flat LDS aperture: addr[31:0] is the LDS offset
}

// ---------------------------------------------------------------------------
// Fast WMMA GEMM, N == 128, K % 32 == 0, M % 16 == 0.
// One block = one 16-row m-tile; 8 waves cover the 8 n-tiles.
// A tile (16x32) staged via double-buffered async global->LDS copies; B is
// pre-packed in fragment order so each lane's 16 B elements are two
// contiguous (coalesced) global_load_b128s.
// Packed B layout: B[((k/32)*8 + c/16)*512 + (c%16)*32 + (k%32)]
// ---------------------------------------------------------------------------
__global__ __launch_bounds__(256) void k_gemm128(
    const unsigned short* __restrict__ A, const unsigned short* __restrict__ Bp,
    const float* __restrict__ bias, float* __restrict__ C, int M, int K)
{
  __shared__ unsigned short sA[2][16 * 32];

  int mtile = blockIdx.x;
  int wid  = threadIdx.x >> 5;
  int lane = threadIdx.x & 31;
  int half = lane >> 4, l16 = lane & 15;

  // per-thread cooperative-copy slot (2 elems = 4 bytes)
  int idx = threadIdx.x * 2;
  int r = idx >> 5, kk = idx & 31;
  const unsigned short* ag = A + (size_t)(mtile * 16 + r) * K + kk;
  unsigned lds0 = lds_off_of(&sA[0][idx]);
  unsigned lds1 = lds_off_of(&sA[1][idx]);

  CFrag c;
#pragma unroll
  for (int i = 0; i < 8; ++i) c.f[i] = 0.f;

  async_ld32(lds0, ag);           // prologue: tile 0 in flight
  int buf = 0;
  for (int k0 = 0; k0 < K; k0 += 32) {
    wait_async0();
    __syncthreads();
    if (k0 + 32 < K)              // prefetch next tile into other buffer
      async_ld32(buf ? lds0 : lds1, ag + k0 + 32);

    AFrag a, b;
    a.q[0] = *(const uint4*)(&sA[buf][l16 * 32 + half * 16]);
    a.q[1] = *(const uint4*)(&sA[buf][l16 * 32 + half * 16 + 8]);
    const uint4* bp = (const uint4*)(Bp + ((size_t)(k0 >> 5) * 8 + wid) * 512
                                        + l16 * 32 + half * 16);
    b.q[0] = bp[0]; b.q[1] = bp[1];
    c.v = __builtin_amdgcn_wmma_f32_16x16x32_bf16(false, a.v, false, b.v,
                                                  (short)0, c.v, false, false);
    buf ^= 1;
  }

  int col = wid * 16 + l16;
  float bv = bias ? bias[col] : 0.f;
#pragma unroll
  for (int r2 = 0; r2 < 8; ++r2) {
    int m = mtile * 16 + half * 8 + r2;
    C[(size_t)m * 128 + col] = c.f[r2] + bv;
  }
}

// ---------------------------------------------------------------------------
// Guarded WMMA GEMM for odd shapes (N=1, N=5), row-major B. Clamped
// unconditional loads + selects (no predicated-load branch chains).
// ---------------------------------------------------------------------------
__global__ __launch_bounds__(256) void k_gemm(
    const unsigned short* __restrict__ A, const unsigned short* __restrict__ B,
    const float* __restrict__ bias, float* __restrict__ C,
    int M, int N, int K)
{
  int ntn  = (N + 15) >> 4;
  int ntm  = (M + 15) >> 4;
  long tile = (long)blockIdx.x * (blockDim.x >> 5) + (threadIdx.x >> 5);
  if (tile >= (long)ntm * ntn) return;
  int tm = (int)(tile / ntn) * 16;
  int tn = (int)(tile % ntn) * 16;
  int lane = threadIdx.x & 31;
  int half = lane >> 4, l16 = lane & 15;
  int row = tm + l16;
  int col = tn + l16;
  int rowc = row < M ? row : M - 1;       // clamped (always in-bounds)
  int colc = col < N ? col : N - 1;
  CFrag c;
#pragma unroll
  for (int i = 0; i < 8; ++i) c.f[i] = 0.f;

  for (int k0 = 0; k0 < K; k0 += 32) {
    AFrag a, b;
#pragma unroll
    for (int i = 0; i < 16; ++i) {
      int k = k0 + half * 16 + i;
      int kc = k < K ? k : K - 1;
      unsigned short va = A[(size_t)rowc * K + kc];
      a.s[i] = (row < M && k < K) ? va : (unsigned short)0;
    }
#pragma unroll
    for (int i = 0; i < 16; ++i) {
      int k = k0 + half * 16 + i;
      int kc = k < K ? k : K - 1;
      unsigned short vb = B[(size_t)kc * N + colc];
      b.s[i] = (k < K && col < N) ? vb : (unsigned short)0;
    }
    c.v = __builtin_amdgcn_wmma_f32_16x16x32_bf16(false, a.v, false, b.v,
                                                  (short)0, c.v, false, false);
  }
#pragma unroll
  for (int r = 0; r < 8; ++r) {
    int m = tm + half * 8 + r;
    if (m < M && col < N) {
      float v = c.f[r];
      if (bias) v += bias[col];
      C[(size_t)m * N + col] = v;
    }
  }
}

// ---------------------------------------------------------------------------
// Fused edge kernel: gathers msg_in = [xin[dst] | xin[src] | ef] on the fly,
// computes m = msg_in @ Bm + bm and a = msg_in @ Ba + ba with ONE shared
// A-fragment per K-step (two WMMAs per step). Block = 16 edges x 128 cols.
// Gather bases hoisted; A staged via double-buffered async global->LDS.
// Bm/Ba fragment-packed (same layout as k_gemm128).
// ---------------------------------------------------------------------------
#define MSG_K 384

__global__ __launch_bounds__(256) void k_msg(
    const unsigned short* __restrict__ xin, const unsigned short* __restrict__ ef,
    const int* __restrict__ srcI, const int* __restrict__ dstI,
    const unsigned short* __restrict__ Bm, const unsigned short* __restrict__ Ba,
    const float* __restrict__ bm, const float* __restrict__ ba,
    float* __restrict__ Om, float* __restrict__ Oa, int E)
{
  __shared__ unsigned short sA[2][16 * 32];

  int etile = blockIdx.x;
  int wid  = threadIdx.x >> 5;       // n-tile 0..7
  int lane = threadIdx.x & 31;
  int half = lane >> 4, l16 = lane & 15;

  // per-thread gather slot: row r of tile, 2 elems at column kk
  int idx = threadIdx.x * 2;
  int r = idx >> 5, kk = idx & 31;
  int e = etile * 16 + r; if (e >= E) e = E - 1;
  const unsigned short* b0 = xin + (size_t)dstI[e] * 128 + kk;   // f in [0,128)
  const unsigned short* b1 = xin + (size_t)srcI[e] * 128 + kk;   // f in [128,256)
  const unsigned short* b2 = ef  + (size_t)e * 128 + kk;         // f in [256,384)
  unsigned lds0 = lds_off_of(&sA[0][idx]);
  unsigned lds1 = lds_off_of(&sA[1][idx]);

  CFrag cm, ca;
#pragma unroll
  for (int i = 0; i < 8; ++i) { cm.f[i] = 0.f; ca.f[i] = 0.f; }

  async_ld32(lds0, b0);            // prologue: k0 = 0 tile
  int buf = 0;
#pragma unroll
  for (int k0 = 0; k0 < MSG_K; k0 += 32) {
    wait_async0();
    __syncthreads();
    int kn = k0 + 32;
    if (kn < MSG_K) {              // prefetch next tile (segment known at compile time)
      int seg = kn >> 7;
      const unsigned short* p = (seg == 0 ? b0 : (seg == 1 ? b1 : b2)) + (kn - (seg << 7));
      async_ld32(buf ? lds0 : lds1, p);
    }

    AFrag a, fbm, fba;
    a.q[0] = *(const uint4*)(&sA[buf][l16 * 32 + half * 16]);
    a.q[1] = *(const uint4*)(&sA[buf][l16 * 32 + half * 16 + 8]);
    size_t boff = ((size_t)(k0 >> 5) * 8 + wid) * 512 + l16 * 32 + half * 16;
    const uint4* bpm = (const uint4*)(Bm + boff);
    const uint4* bpa = (const uint4*)(Ba + boff);
    fbm.q[0] = bpm[0]; fbm.q[1] = bpm[1];
    fba.q[0] = bpa[0]; fba.q[1] = bpa[1];
    cm.v = __builtin_amdgcn_wmma_f32_16x16x32_bf16(false, a.v, false, fbm.v,
                                                   (short)0, cm.v, false, false);
    ca.v = __builtin_amdgcn_wmma_f32_16x16x32_bf16(false, a.v, false, fba.v,
                                                   (short)0, ca.v, false, false);
    buf ^= 1;
  }

  int col = wid * 16 + l16;
#pragma unroll
  for (int r2 = 0; r2 < 8; ++r2) {
    int eo = etile * 16 + half * 8 + r2;
    if (eo < E) {
      Om[(size_t)eo * 128 + col] = cm.f[r2] + bm[col];
      Oa[(size_t)eo * 128 + col] = ca.f[r2] + ba[col];
    }
  }
}

// ---------------------------------------------------------------------------
// LayerNorm + SiLU, one wave per normalized row of `group` (32 or 128) elems.
// gamma index = (row % gmod)*group + j  (gmod=4 for per-head LN, else 1)
// ---------------------------------------------------------------------------
__global__ __launch_bounds__(256) void k_ln_silu(
    float* __restrict__ x, const float* __restrict__ g, const float* __restrict__ b,
    long nrows, int group, int gmod)
{
  long row = ((long)blockIdx.x * blockDim.x + threadIdx.x) >> 5;
  if (row >= nrows) return;
  int lane = threadIdx.x & 31;
  int ep = group >> 5;              // 1 or 4
  float* r = x + row * group;
  float v[4]; float s = 0.f;
  for (int i = 0; i < ep; ++i) { v[i] = r[lane + 32 * i]; s += v[i]; }
  for (int m = 16; m > 0; m >>= 1) s += __shfl_xor(s, m, 32);
  float mean = s / group;
  float var = 0.f;
  for (int i = 0; i < ep; ++i) { float d = v[i] - mean; var += d * d; }
  for (int m = 16; m > 0; m >>= 1) var += __shfl_xor(var, m, 32);
  float inv = rsqrtf(var / group + 1e-5f);
  int gbase = (int)(row % gmod) * group;
  for (int i = 0; i < ep; ++i) {
    int c = lane + 32 * i;
    float y = (v[i] - mean) * inv * g[gbase + c] + b[gbase + c];
    r[c] = y / (1.f + expf(-y));    // SiLU
  }
}

// ---------------------------------------------------------------------------
// Small utility kernels
// ---------------------------------------------------------------------------
__global__ void k_zero(float* p, long n)               { long t=(long)blockIdx.x*blockDim.x+threadIdx.x; if(t<n) p[t]=0.f; }
__global__ void k_fill_u32(unsigned* p, unsigned v, long n){ long t=(long)blockIdx.x*blockDim.x+threadIdx.x; if(t<n) p[t]=v; }
__global__ void k_copyf(float* d, const float* s, long n){ long t=(long)blockIdx.x*blockDim.x+threadIdx.x; if(t<n) d[t]=s[t]; }
__global__ void k_cast(unsigned short* d, const float* s, long n){ long t=(long)blockIdx.x*blockDim.x+threadIdx.x; if(t<n) d[t]=f2bf(s[t]); }
__global__ void k_addinto(float* d, const float* s, long n){ long t=(long)blockIdx.x*blockDim.x+threadIdx.x; if(t<n) d[t]+=s[t]; }

// cast with row padding: src [rows,kin] -> dst [rows,kout], tail already zeroed
__global__ void k_cast_pad(unsigned short* dst, const float* src, long n, int kin, int kout) {
  long t = (long)blockIdx.x * blockDim.x + threadIdx.x; if (t >= n) return;
  long r = t / kin; int c = (int)(t % kin);
  dst[r * kout + c] = f2bf(src[t]);
}

// pack row-major f32 B[kin,128] -> bf16 fragment order, zero-padded to K rows
// dst[((k/32)*8 + c/16)*512 + (c%16)*32 + (k%32)]
__global__ void k_packB(unsigned short* dst, const float* src, long n, int kin) {
  long t = (long)blockIdx.x * blockDim.x + threadIdx.x; if (t >= n) return; // n = K*128
  int k = (int)(t >> 7), c = (int)(t & 127);
  float v = (k < kin) ? src[(size_t)k * 128 + c] : 0.f;
  long di = ((long)((k >> 5) * 8 + (c >> 4))) * 512 + (c & 15) * 32 + (k & 31);
  dst[di] = f2bf(v);
}

// pack msg/att weights (H,F,D)=(4,384,32): out col c = h*32+d -> fragment order
__global__ void k_packB_msg(unsigned short* dst, const float* src, long n) { // n = 384*128
  long t = (long)blockIdx.x * blockDim.x + threadIdx.x; if (t >= n) return;
  int f = (int)(t >> 7), c = (int)(t & 127);
  float v = src[(size_t)(c >> 5) * 384 * 32 + (size_t)f * 32 + (c & 31)];
  long di = ((long)((f >> 5) * 8 + (c >> 4))) * 512 + (c & 15) * 32 + (f & 31);
  dst[di] = f2bf(v);
}

__global__ void k_make_xin(unsigned short* xin, const float* h, const float* vn, long n) {
  long t = (long)blockIdx.x * blockDim.x + threadIdx.x; if (t >= n) return;
  xin[t] = f2bf(h[t] + vn[t & 127]);
}

__global__ void k_att(const float* __restrict__ a, const float* __restrict__ w2,
                      const float* __restrict__ b2, float* __restrict__ att, long EH) {
  long t = (long)blockIdx.x * blockDim.x + threadIdx.x; if (t >= EH) return;
  int h = (int)(t & 3); long e = t >> 2;
  const float* ar = a + e * 128 + h * 32;
  const float* wr = w2 + h * 32;
  float s = b2[h];
#pragma unroll
  for (int d = 0; d < 32; ++d) s += ar[d] * wr[d];
  att[t] = 1.f / (1.f + expf(-s));
}

__global__ void k_scatter_aggr(const float* __restrict__ m, const float* __restrict__ att,
                               const int* __restrict__ dstI, float* __restrict__ aggr, long n) {
  long t = (long)blockIdx.x * blockDim.x + threadIdx.x; if (t >= n) return;
  long e = t >> 7; int c = (int)(t & 127); int h = c >> 5;
  atomicAdd(&aggr[(size_t)dstI[e] * 128 + c], m[t] * att[e * 4 + h]);
}

__global__ void k_add_bcast(float* __restrict__ upd, const float* __restrict__ xwx, long n) {
  long t = (long)blockIdx.x * blockDim.x + threadIdx.x; if (t >= n) return;
  long nrow = t >> 9; int c = (int)(t & 127);
  upd[t] += xwx[nrow * 128 + c];
}

__global__ void k_count(const int* batch, float* cnt, long n) {
  long t = (long)blockIdx.x * blockDim.x + threadIdx.x; if (t >= n) return;
  atomicAdd(&cnt[batch[t]], 1.f);
}

__global__ void k_scatter_nodes(const float* __restrict__ x, const int* __restrict__ batch,
                                float* __restrict__ vng, long n) {
  long t = (long)blockIdx.x * blockDim.x + threadIdx.x; if (t >= n) return;
  atomicAdd(&vng[(size_t)batch[t >> 7] * 128 + (t & 127)], x[t]);
}

__global__ void k_vn_final(const float* vng, const float* cnt, float* vn, int G) {
  int c = threadIdx.x; if (c >= 128) return;
  float s = 0.f;
  for (int g = 0; g < G; ++g) s += vng[(size_t)g * 128 + c] / fmaxf(cnt[g], 1.f);
  vn[c] = s / (float)G;
}

__global__ void k_segmax(const float* pw, const int* batch, unsigned* mx, long n) {
  long t = (long)blockIdx.x * blockDim.x + threadIdx.x; if (t >= n) return;
  atomicMax(&mx[batch[t]], fenc(pw[t]));
}

__global__ void k_expw(const float* pw, const int* batch, const unsigned* mx,
                       float* wgt, float* s, long n) {
  long t = (long)blockIdx.x * blockDim.x + threadIdx.x; if (t >= n) return;
  int g = batch[t];
  float e = expf(pw[t] - fdec(mx[g]));
  wgt[t] = e;
  atomicAdd(&s[g], e);
}

__global__ void k_pool_scatter(const float* __restrict__ h, const int* __restrict__ batch,
                               const float* __restrict__ wgt, const float* __restrict__ s,
                               float* __restrict__ gvec, long n) {
  long t = (long)blockIdx.x * blockDim.x + threadIdx.x; if (t >= n) return;
  long nn = t >> 7; int c = (int)(t & 127); int g = batch[nn];
  atomicAdd(&gvec[(size_t)g * 128 + c], h[t] * wgt[nn] / s[g]);
}

// ---------------------------------------------------------------------------
// Host orchestration
// ---------------------------------------------------------------------------
extern "C" void kernel_launch(void* const* d_in, const int* in_sizes, int n_in,
                              void* d_out, int out_size, void* d_ws, size_t ws_size,
                              hipStream_t stream) {
  const int TASKS = 5, G = 128, LAYERS = 3;
  const int N = in_sizes[0] / 64;        // 40000
  const int E = in_sizes[2] / 16;        // 256000

  const float* x     = (const float*)d_in[0];
  const int*   ei    = (const int*)d_in[1];
  const int*   srcI  = ei;
  const int*   dstI  = ei + E;
  const float* ea    = (const float*)d_in[2];
  const int*   batch = (const int*)d_in[3];
  auto pf = [&](int i) { return (const float*)d_in[i]; };
  // params: 5=embed.w 6=embed.b ; layers start at 7, 23 arrays each ; tail at 76
  const int LB = 7, TAIL = 76;

  // ---- workspace bump allocator (256B aligned) ----
  char* base = (char*)d_ws; size_t off = 0;
  auto alloc = [&](size_t bytes) { off = (off + 255) & ~(size_t)255; void* p = base + off; off += bytes; return p; };
  auto allocF = [&](long n) { return (float*)alloc((size_t)n * 4); };
  auto allocH = [&](long n) { return (unsigned short*)alloc((size_t)n * 2); };

  unsigned short* embW = allocH(64 * 128);         // fragment-packed
  unsigned short *elW[3], *msgB[3], *attB[3], *waB[3], *wxB[3], *outW[3];
  for (int l = 0; l < LAYERS; ++l) {
    elW[l]  = allocH(32 * 128);          // K padded 16 -> 32, packed
    msgB[l] = allocH(384 * 128); attB[l] = allocH(384 * 128);   // packed
    waB[l]  = allocH(32 * 128);  wxB[l]  = allocH(128 * 128); outW[l] = allocH(512 * 128);
  }
  unsigned short* poolW1 = allocH(128 * 128);      // packed
  unsigned short* poolW2 = allocH(128 * 1);        // row-major (guarded path)
  unsigned short* meanW1 = allocH(128 * 128);      // packed
  unsigned short* meanW2 = allocH(128 * TASKS);    // row-major (guarded path)
  unsigned short* lvW1   = allocH(128 * 128);      // packed
  unsigned short* lvW2   = allocH(128 * TASKS);    // row-major (guarded path)

  unsigned short* x_bf   = allocH((long)N * 64);
  unsigned short* ea_bf  = allocH((long)E * 32);   // K padded 16 -> 32
  unsigned short* xin_bf = allocH((long)N * 128);
  unsigned short* ef_bf  = allocH((long)E * 128);
  unsigned short* agg_bf = allocH((long)N * 128);
  unsigned short* hbf    = allocH((long)N * 128);
  float* h    = allocF((long)N * 128);
  float* vn   = allocF(128);
  float* vng  = allocF((long)G * 128);
  float* cnt  = allocF(G);
  float* ef32 = allocF((long)E * 128);   // ef -> m -> (as ushort) comb_bf
  float* a32  = allocF((long)E * 128);   // a  -> upd
  float* att  = allocF((long)E * 4);
  float* aggr = allocF((long)N * 128);
  float* xwx  = allocF((long)N * 128);   // also reused for pool t
  float* xupd = allocF((long)N * 128);
  float* pw   = allocF(N);
  float* wgt  = allocF(N);
  float* ssum = allocF(G);
  unsigned* mxu = (unsigned*)alloc((size_t)G * 4);
  float* gvec = allocF((long)G * 128);
  unsigned short* g_bf  = allocH((long)G * 128);
  float* t2   = allocF((long)G * 128);
  unsigned short* t2_bf = allocH((long)G * 128);
  (void)ws_size;

  auto blks = [](long n) { return (int)((n + 255) / 256); };
  auto zero = [&](float* p, long n) { k_zero<<<blks(n), 256, 0, stream>>>(p, n); };
  auto cast = [&](unsigned short* d, const float* s, long n) { k_cast<<<blks(n), 256, 0, stream>>>(d, s, n); };
  auto packB = [&](unsigned short* d, const float* s, int K, int kin) {
    k_packB<<<blks((long)K * 128), 256, 0, stream>>>(d, s, (long)K * 128, kin);
  };
  auto gemm128 = [&](const unsigned short* A, const unsigned short* Bp, const float* bias,
                     float* C, int M, int K) {
    k_gemm128<<<M / 16, 256, 0, stream>>>(A, Bp, bias, C, M, K);
  };
  auto gemmg = [&](const unsigned short* A, const unsigned short* B, const float* bias,
                   float* C, int M, int Nn, int K) {
    long waves = (long)((M + 15) / 16) * ((Nn + 15) / 16);
    k_gemm<<<(int)((waves + 7) / 8), 256, 0, stream>>>(A, B, bias, C, M, Nn, K);
  };
  auto lnsilu = [&](float* p, const float* g, const float* b, long rows, int group, int gmod) {
    k_ln_silu<<<blks(rows * 32), 256, 0, stream>>>(p, g, b, rows, group, gmod);
  };

  // ---- weight prep (bf16, fragment-packed for fast path) ----
  packB(embW, pf(5), 64, 64);
  for (int l = 0; l < LAYERS; ++l) {
    int b = LB + l * 23;
    packB(elW[l], pf(b + 0), 32, 16);    // zero-padded K 16 -> 32
    k_packB_msg<<<blks(384 * 128), 256, 0, stream>>>(msgB[l], pf(b + 4), 384 * 128);
    k_packB_msg<<<blks(384 * 128), 256, 0, stream>>>(attB[l], pf(b + 8), 384 * 128);
    packB(waB[l], pf(b + 14), 32, 32);
    packB(wxB[l], pf(b + 15), 128, 128);
    packB(outW[l], pf(b + 19), 512, 512);
  }
  packB(poolW1, pf(TAIL + 0), 128, 128);
  cast(poolW2, pf(TAIL + 4), 128 * 1);
  packB(meanW1, pf(TAIL + 6), 128, 128);
  cast(meanW2, pf(TAIL + 10), 128 * TASKS);
  packB(lvW1, pf(TAIL + 12), 128, 128);
  cast(lvW2, pf(TAIL + 16), 128 * TASKS);

  cast(x_bf, x, (long)N * 64);
  k_fill_u32<<<blks((long)E * 32 / 2), 256, 0, stream>>>((unsigned*)ea_bf, 0u, (long)E * 32 / 2);
  k_cast_pad<<<blks((long)E * 16), 256, 0, stream>>>(ea_bf, ea, (long)E * 16, 16, 32);
  k_copyf<<<1, 128, 0, stream>>>(vn, pf(TAIL + 18), 128);   // params["vn"]

  zero(cnt, G);
  k_count<<<blks(N), 256, 0, stream>>>(batch, cnt, N);

  // ---- embed ----
  gemm128(x_bf, embW, pf(6), h, N, 64);

  // ---- layers ----
  for (int l = 0; l < LAYERS; ++l) {
    int b = LB + l * 23;
    // edge features (K padded to 32)
    gemm128(ea_bf, elW[l], pf(b + 1), ef32, E, 32);
    lnsilu(ef32, pf(b + 2), pf(b + 3), E, 128, 1);
    cast(ef_bf, ef32, (long)E * 128);
    // xin = h + vn (bf16)
    k_make_xin<<<blks((long)N * 128), 256, 0, stream>>>(xin_bf, h, vn, (long)N * 128);
    // fused gather + dual GEMM (m into ef32, a into a32)
    k_msg<<<E / 16, 256, 0, stream>>>(xin_bf, ef_bf, srcI, dstI, msgB[l], attB[l],
                                      pf(b + 5), pf(b + 9), ef32, a32, E);
    lnsilu(ef32, pf(b + 6), pf(b + 7), (long)E * 4, 32, 4);    // m per-head LN+SiLU
    lnsilu(a32,  pf(b + 10), pf(b + 11), (long)E * 4, 32, 4);  // a per-head LN+SiLU
    k_att<<<blks((long)E * 4), 256, 0, stream>>>(a32, pf(b + 12), pf(b + 13), att, (long)E * 4);
    // aggregate messages
    zero(aggr, (long)N * 128);
    k_scatter_aggr<<<blks((long)E * 128), 256, 0, stream>>>(ef32, att, dstI, aggr, (long)E * 128);
    cast(agg_bf, aggr, (long)N * 128);
    // update MLP
    gemm128(xin_bf, wxB[l], pf(b + 16), xwx, N, 128);          // xin@wx + upd_b
    float* upd = a32;                                          // reuse [N*4,128]
    gemm128(agg_bf, waB[l], nullptr, upd, N * 4, 32);          // aggr@wa
    k_add_bcast<<<blks((long)N * 512), 256, 0, stream>>>(upd, xwx, (long)N * 512);
    lnsilu(upd, pf(b + 17), pf(b + 18), (long)N * 4, 128, 1);
    unsigned short* comb_bf = (unsigned short*)ef32;           // reuse
    cast(comb_bf, upd, (long)N * 512);
    gemm128(comb_bf, outW[l], pf(b + 20), xupd, N, 512);
    lnsilu(xupd, pf(b + 21), pf(b + 22), N, 128, 1);
    // virtual node (skip on last layer: unused)
    if (l + 1 < LAYERS) {
      zero(vng, (long)G * 128);
      k_scatter_nodes<<<blks((long)N * 128), 256, 0, stream>>>(xupd, batch, vng, (long)N * 128);
      k_vn_final<<<1, 128, 0, stream>>>(vng, cnt, vn, G);
    }
    k_addinto<<<blks((long)N * 128), 256, 0, stream>>>(h, xupd, (long)N * 128);
  }

  // ---- attention pooling ----
  cast(hbf, h, (long)N * 128);
  float* t = xwx;                                  // reuse
  gemm128(hbf, poolW1, pf(TAIL + 1), t, N, 128);
  lnsilu(t, pf(TAIL + 2), pf(TAIL + 3), N, 128, 1);
  unsigned short* t_bf = agg_bf;                   // reuse
  cast(t_bf, t, (long)N * 128);
  gemmg(t_bf, poolW2, pf(TAIL + 5), pw, N, 1, 128);
  k_fill_u32<<<1, G, 0, stream>>>(mxu, FENC_NEG_INF, G);
  k_segmax<<<blks(N), 256, 0, stream>>>(pw, batch, mxu, N);
  zero(ssum, G);
  k_expw<<<blks(N), 256, 0, stream>>>(pw, batch, mxu, wgt, ssum, N);
  zero(gvec, (long)G * 128);
  k_pool_scatter<<<blks((long)N * 128), 256, 0, stream>>>(h, batch, wgt, ssum, gvec, (long)N * 128);
  cast(g_bf, gvec, (long)G * 128);

  // ---- mean / logvar heads ----
  float* out = (float*)d_out;
  gemm128(g_bf, meanW1, pf(TAIL + 7), t2, G, 128);
  lnsilu(t2, pf(TAIL + 8), pf(TAIL + 9), G, 128, 1);
  cast(t2_bf, t2, (long)G * 128);
  gemmg(t2_bf, meanW2, pf(TAIL + 11), out, G, TASKS, 128);
  gemm128(g_bf, lvW1, pf(TAIL + 13), t2, G, 128);
  lnsilu(t2, pf(TAIL + 14), pf(TAIL + 15), G, 128, 1);
  cast(t2_bf, t2, (long)G * 128);
  gemmg(t2_bf, lvW2, pf(TAIL + 17), out + G * TASKS, G, TASKS, 128);
}
